// GRU_68247030334406
// MI455X (gfx1250) — matrix operand order repaired and checked by reading, not compiled, not run on hardware
//
#include <hip/hip_runtime.h>
#include <hip/hip_bf16.h>

// Problem constants (match reference)
#define S_ 64
#define B_ 64
#define H_ 1024
#define E_ 512
#define V_ 32000
#define L_ 2

typedef __attribute__((ext_vector_type(16))) __bf16 v16bf;
typedef __attribute__((ext_vector_type(8)))  __bf16 v8bf;
typedef __attribute__((ext_vector_type(8)))  float  v8f;

union V16U { v16bf v; v8bf h[2]; };

// ---------------------------------------------------------------------------
// WMMA helper (CDNA5 / gfx1250, wave32, v_wmma_f32_16x16x32_bf16)
// ---------------------------------------------------------------------------
__device__ __forceinline__ v8f wmma_bf16(v16bf a, v16bf b, v8f c) {
  // 8 args: (neg_a, A, neg_b, B, c_mod, C, reuse_a, reuse_b)
  return __builtin_amdgcn_wmma_f32_16x16x32_bf16(false, a, false, b, (short)0, c, false, false);
}

// Load a 16x32 (MxK) A-tile from a row-major bf16 matrix.
// ISA layout: lane l -> row m = l&15; kg = l>>4;
//   elements 0..7  hold K = 8*kg + {0..7}
//   elements 8..15 hold K = 16 + 8*kg + {0..7}
// Two 16-byte loads straight into the WMMA source VGPRs.
__device__ __forceinline__ v16bf load_A_bf(const __bf16* __restrict__ src, int ld,
                                           int m0, int k0) {
  int lane = threadIdx.x & 31;
  int m = lane & 15, kg = lane >> 4;
  const __bf16* p = src + (size_t)(m0 + m) * ld + k0 + 8 * kg;
  V16U u;
  u.h[0] = *(const v8bf*)p;          // K = 8*kg + 0..7
  u.h[1] = *(const v8bf*)(p + 16);   // K = 16 + 8*kg + 0..7
  return u.v;
}

// Load a 32x16 (KxN) B-tile where B[k][n] = W[n0+n][k0+k], W row-major [N][K] bf16.
// ISA layout: lane l -> col n = l&15; kg = l>>4; elements 0..7 hold K = 8*kg + {0..7}, 8..15 hold K = 16 + 8*kg + {0..7}.
__device__ __forceinline__ v16bf load_B_bf(const __bf16* __restrict__ W, int ld,
                                           int n0, int k0) {
  int lane = threadIdx.x & 31;
  int n = lane & 15, kg = lane >> 4;
  const __bf16* p = W + (size_t)(n0 + n) * ld + k0 + 8 * kg;
  V16U u;
  u.h[0] = *(const v8bf*)p;
  u.h[1] = *(const v8bf*)(p + 16);
  return u.v;
}

__device__ __forceinline__ float sigmoidf_(float x) {
  return 1.0f / (1.0f + __expf(-x));
}

// ---------------------------------------------------------------------------
// f32 -> bf16 bulk conversion (8 elements / thread, vector loads+stores)
// ---------------------------------------------------------------------------
__global__ void cvt_bf16_kernel(const float* __restrict__ src,
                                __bf16* __restrict__ dst, size_t n8) {
  size_t i = (size_t)blockIdx.x * blockDim.x + threadIdx.x;
  if (i >= n8) return;
  const float4* s = (const float4*)src + i * 2;
  float4 a = s[0], b = s[1];
  v8bf o;
  o[0] = (__bf16)a.x; o[1] = (__bf16)a.y; o[2] = (__bf16)a.z; o[3] = (__bf16)a.w;
  o[4] = (__bf16)b.x; o[5] = (__bf16)b.y; o[6] = (__bf16)b.z; o[7] = (__bf16)b.w;
  *((v8bf*)dst + i) = o;
}

// ---------------------------------------------------------------------------
// Prologue: gather embeddings as bf16, init h0/h1 (f32 + bf16 mirrors),
// copy pass-through hidden to output tail.
// ---------------------------------------------------------------------------
__global__ void prologue_kernel(const int* __restrict__ idx,
                                const float* __restrict__ emb_table,
                                __bf16* __restrict__ xall,        // [S,B,E] bf16
                                const float* __restrict__ hidden, // [L,B,H]
                                float* __restrict__ h0f, float* __restrict__ h1f,
                                __bf16* __restrict__ h0b, __bf16* __restrict__ h1b,
                                float* __restrict__ out_tail) {
  size_t i = (size_t)blockIdx.x * blockDim.x + threadIdx.x;
  size_t total = (size_t)S_ * B_ * E_;
  if (i < total) {
    size_t sb = i / E_, e = i - sb * E_;
    xall[i] = (__bf16)emb_table[(size_t)idx[sb] * E_ + e];
  }
  if (i < (size_t)B_ * H_) {
    float a = hidden[i];
    float b = hidden[(size_t)B_ * H_ + i];
    h0f[i] = a; h0b[i] = (__bf16)a;
    h1f[i] = b; h1b[i] = (__bf16)b;
  }
  if (i < (size_t)L_ * B_ * H_) out_tail[i] = hidden[i];
}

// ---------------------------------------------------------------------------
// Fused r/z gate kernel for one GRU cell at one timestep.
//   r = sigmoid(x@Wr^T + bWr + h@Ur^T + bUr)
//   z = sigmoid(x@Wz^T + bWz + h@Uz^T + bUz)
// Writes rh = r*h (bf16, next matmul input) and z (f32).
// One wave per (16 x 32) output tile; A-tiles shared across both gates.
// Grid: 4 m-tiles * 32 n-tiles = 128 waves -> <<<32,128>>>
// ---------------------------------------------------------------------------
template <int KX>
__global__ void gate_rz_kernel(const __bf16* __restrict__ x,    // [B_,KX] bf16
                               const __bf16* __restrict__ hb,   // [B_,H_] bf16
                               const float*  __restrict__ hf,   // [B_,H_] f32
                               const __bf16* __restrict__ Wr, const float* __restrict__ bWr,
                               const __bf16* __restrict__ Wz, const float* __restrict__ bWz,
                               const __bf16* __restrict__ Ur, const float* __restrict__ bUr,
                               const __bf16* __restrict__ Uz, const float* __restrict__ bUz,
                               __bf16* __restrict__ rhb, float* __restrict__ zbuf) {
  int wave = (int)((blockIdx.x * blockDim.x + threadIdx.x) >> 5);
  int mt = wave & 3;
  int nt = wave >> 2;
  int m0 = mt * 16, n0 = nt * 32;

  v8f ar0 = {}, ar1 = {}, az0 = {}, az1 = {};

#pragma unroll 4
  for (int k = 0; k < KX; k += 32) {            // input-to-hidden part
    v16bf a = load_A_bf(x, KX, m0, k);
    ar0 = wmma_bf16(a, load_B_bf(Wr, KX, n0,      k), ar0);
    ar1 = wmma_bf16(a, load_B_bf(Wr, KX, n0 + 16, k), ar1);
    az0 = wmma_bf16(a, load_B_bf(Wz, KX, n0,      k), az0);
    az1 = wmma_bf16(a, load_B_bf(Wz, KX, n0 + 16, k), az1);
  }
#pragma unroll 4
  for (int k = 0; k < H_; k += 32) {            // hidden-to-hidden part
    v16bf a = load_A_bf(hb, H_, m0, k);
    ar0 = wmma_bf16(a, load_B_bf(Ur, H_, n0,      k), ar0);
    ar1 = wmma_bf16(a, load_B_bf(Ur, H_, n0 + 16, k), ar1);
    az0 = wmma_bf16(a, load_B_bf(Uz, H_, n0,      k), az0);
    az1 = wmma_bf16(a, load_B_bf(Uz, H_, n0 + 16, k), az1);
  }

  int lane = threadIdx.x & 31;
  int nl = lane & 15, kg = lane >> 4;
#pragma unroll
  for (int v = 0; v < 8; ++v) {
    int m = m0 + v + 8 * kg;
    {
      int n = n0 + nl;
      float r = sigmoidf_(ar0[v] + bWr[n] + bUr[n]);
      float z = sigmoidf_(az0[v] + bWz[n] + bUz[n]);
      rhb[(size_t)m * H_ + n] = (__bf16)(r * hf[(size_t)m * H_ + n]);
      zbuf[(size_t)m * H_ + n] = z;
    }
    {
      int n = n0 + 16 + nl;
      float r = sigmoidf_(ar1[v] + bWr[n] + bUr[n]);
      float z = sigmoidf_(az1[v] + bWz[n] + bUz[n]);
      rhb[(size_t)m * H_ + n] = (__bf16)(r * hf[(size_t)m * H_ + n]);
      zbuf[(size_t)m * H_ + n] = z;
    }
  }
}

// ---------------------------------------------------------------------------
// h_tilde + state update:
//   h_tilde = tanh(x@Wh^T + bWh + rh@Uh^T + bUh)
//   h = (1-z)*h + z*h_tilde      (writes f32 state + bf16 mirror)
// ---------------------------------------------------------------------------
template <int KX>
__global__ void htilde_kernel(const __bf16* __restrict__ x,    // [B_,KX] bf16
                              const __bf16* __restrict__ rhb,  // [B_,H_] bf16
                              const __bf16* __restrict__ Wh, const float* __restrict__ bWh,
                              const __bf16* __restrict__ Uh, const float* __restrict__ bUh,
                              const float* __restrict__ zbuf,
                              float* __restrict__ hf, __bf16* __restrict__ hb) {
  int wave = (int)((blockIdx.x * blockDim.x + threadIdx.x) >> 5);
  int mt = wave & 3;
  int nt = wave >> 2;
  int m0 = mt * 16, n0 = nt * 32;

  v8f a0 = {}, a1 = {};
#pragma unroll 4
  for (int k = 0; k < KX; k += 32) {
    v16bf a = load_A_bf(x, KX, m0, k);
    a0 = wmma_bf16(a, load_B_bf(Wh, KX, n0,      k), a0);
    a1 = wmma_bf16(a, load_B_bf(Wh, KX, n0 + 16, k), a1);
  }
#pragma unroll 4
  for (int k = 0; k < H_; k += 32) {
    v16bf a = load_A_bf(rhb, H_, m0, k);
    a0 = wmma_bf16(a, load_B_bf(Uh, H_, n0,      k), a0);
    a1 = wmma_bf16(a, load_B_bf(Uh, H_, n0 + 16, k), a1);
  }

  int lane = threadIdx.x & 31;
  int nl = lane & 15, kg = lane >> 4;
#pragma unroll
  for (int v = 0; v < 8; ++v) {
    int m = m0 + v + 8 * kg;
    {
      int n = n0 + nl;
      float ht = tanhf(a0[v] + bWh[n] + bUh[n]);
      float z = zbuf[(size_t)m * H_ + n];
      float hv = hf[(size_t)m * H_ + n];
      float hn = (1.0f - z) * hv + z * ht;
      hf[(size_t)m * H_ + n] = hn;
      hb[(size_t)m * H_ + n] = (__bf16)hn;
    }
    {
      int n = n0 + 16 + nl;
      float ht = tanhf(a1[v] + bWh[n] + bUh[n]);
      float z = zbuf[(size_t)m * H_ + n];
      float hv = hf[(size_t)m * H_ + n];
      float hn = (1.0f - z) * hv + z * ht;
      hf[(size_t)m * H_ + n] = hn;
      hb[(size_t)m * H_ + n] = (__bf16)hn;
    }
  }
}

// ---------------------------------------------------------------------------
// Decoder: logits = h1 @ dec_W^T + dec_b   (64 x 32000, K=1024)
// One wave owns a 64x32 output tile (all 4 M-tiles in registers -> dec_W bf16
// is streamed exactly once per step). 8 WMMAs per K-iteration.
// Grid: 1000 waves -> <<<250,128>>>
// ---------------------------------------------------------------------------
__global__ void decode_kernel(const __bf16* __restrict__ h1b,   // [B_,H_] bf16
                              const __bf16* __restrict__ decW,  // [V_,H_] bf16
                              const float* __restrict__ decb,
                              float* __restrict__ out) {        // [B_,V_] this step
  int wave = (int)((blockIdx.x * blockDim.x + threadIdx.x) >> 5);
  int n0 = wave * 32;

  v8f acc[4][2] = {};
#pragma unroll 2
  for (int k = 0; k < H_; k += 32) {
    v16bf b0 = load_B_bf(decW, H_, n0,      k);
    v16bf b1 = load_B_bf(decW, H_, n0 + 16, k);
#pragma unroll
    for (int mt = 0; mt < 4; ++mt) {
      v16bf a = load_A_bf(h1b, H_, mt * 16, k);
      acc[mt][0] = wmma_bf16(a, b0, acc[mt][0]);
      acc[mt][1] = wmma_bf16(a, b1, acc[mt][1]);
    }
  }

  int lane = threadIdx.x & 31;
  int nl = lane & 15, kg = lane >> 4;
#pragma unroll
  for (int mt = 0; mt < 4; ++mt) {
#pragma unroll
    for (int v = 0; v < 8; ++v) {
      int m = mt * 16 + v + 8 * kg;
      int n = n0 + nl;
      out[(size_t)m * V_ + n] = acc[mt][0][v] + decb[n];
      n = n0 + 16 + nl;
      out[(size_t)m * V_ + n] = acc[mt][1][v] + decb[n];
    }
  }
}

// ---------------------------------------------------------------------------
// Host-side launch
// ---------------------------------------------------------------------------
static inline void cvt(const float* src, __bf16* dst, size_t n, hipStream_t stream) {
  size_t n8 = n / 8;  // all sizes here are multiples of 8
  int blocks = (int)((n8 + 255) / 256);
  cvt_bf16_kernel<<<blocks, 256, 0, stream>>>(src, dst, n8);
}

extern "C" void kernel_launch(void* const* d_in, const int* in_sizes, int n_in,
                              void* d_out, int out_size, void* d_ws, size_t ws_size,
                              hipStream_t stream) {
  const int*   inputs    = (const int*)d_in[0];
  const float* hidden    = (const float*)d_in[1];
  const float* emb_table = (const float*)d_in[2];
  const float* dec_W     = (const float*)d_in[3];
  const float* dec_b     = (const float*)d_in[4];

  // Per-layer weights, dict order: Wr,bWr,Wz,bWz,Wh,bWh,Ur,bUr,Uz,bUz,Uh,bUh
  const float* Wp[2][12];
  for (int l = 0; l < 2; ++l)
    for (int j = 0; j < 12; ++j)
      Wp[l][j] = (const float*)d_in[5 + l * 12 + j];

  float* out = (float*)d_out;
  float* out_tail = out + (size_t)S_ * B_ * V_;  // pass-through hidden

  // ---------------- Workspace layout ----------------
  char* wsp = (char*)d_ws;
  auto take = [&](size_t bytes) { char* p = wsp; wsp += (bytes + 255) & ~(size_t)255; return p; };

  // f32 buffers
  float* h0f = (float*)take((size_t)B_ * H_ * 4);
  float* h1f = (float*)take((size_t)B_ * H_ * 4);
  float* zb  = (float*)take((size_t)B_ * H_ * 4);
  // bf16 buffers
  __bf16* xall = (__bf16*)take((size_t)S_ * B_ * E_ * 2);
  __bf16* h0b  = (__bf16*)take((size_t)B_ * H_ * 2);
  __bf16* h1b  = (__bf16*)take((size_t)B_ * H_ * 2);
  __bf16* rhb  = (__bf16*)take((size_t)B_ * H_ * 2);
  __bf16* decWb = (__bf16*)take((size_t)V_ * H_ * 2);
  __bf16* Wb[2][6];  // bf16 copies of Wr,Wz,Wh,Ur,Uz,Uh per layer
  for (int l = 0; l < 2; ++l) {
    size_t din = (l == 0) ? E_ : H_;
    Wb[l][0] = (__bf16*)take((size_t)H_ * din * 2);  // Wr
    Wb[l][1] = (__bf16*)take((size_t)H_ * din * 2);  // Wz
    Wb[l][2] = (__bf16*)take((size_t)H_ * din * 2);  // Wh
    Wb[l][3] = (__bf16*)take((size_t)H_ * H_ * 2);   // Ur
    Wb[l][4] = (__bf16*)take((size_t)H_ * H_ * 2);   // Uz
    Wb[l][5] = (__bf16*)take((size_t)H_ * H_ * 2);   // Uh
  }

  // ---------------- Weight conversion (once per launch) ----------------
  cvt(dec_W, decWb, (size_t)V_ * H_, stream);
  for (int l = 0; l < 2; ++l) {
    size_t din = (l == 0) ? E_ : H_;
    cvt(Wp[l][0], Wb[l][0], (size_t)H_ * din, stream);  // Wr
    cvt(Wp[l][2], Wb[l][1], (size_t)H_ * din, stream);  // Wz
    cvt(Wp[l][4], Wb[l][2], (size_t)H_ * din, stream);  // Wh
    cvt(Wp[l][6], Wb[l][3], (size_t)H_ * H_, stream);   // Ur
    cvt(Wp[l][8], Wb[l][4], (size_t)H_ * H_, stream);   // Uz
    cvt(Wp[l][10], Wb[l][5], (size_t)H_ * H_, stream);  // Uh
  }

  // ---------------- Prologue ----------------
  {
    size_t total = (size_t)S_ * B_ * E_;
    int blocks = (int)((total + 255) / 256);
    prologue_kernel<<<blocks, 256, 0, stream>>>(inputs, emb_table, xall, hidden,
                                                h0f, h1f, h0b, h1b, out_tail);
  }

  // ---------------- Recurrence ----------------
  for (int s = 0; s < S_; ++s) {
    const __bf16* xs = xall + (size_t)s * B_ * E_;

    // Layer 0 (x = embedding, Kx = E)
    gate_rz_kernel<E_><<<32, 128, 0, stream>>>(xs, h0b, h0f,
        Wb[0][0], Wp[0][1], Wb[0][1], Wp[0][3],
        Wb[0][3], Wp[0][7], Wb[0][4], Wp[0][9],
        rhb, zb);
    htilde_kernel<E_><<<32, 128, 0, stream>>>(xs, rhb,
        Wb[0][2], Wp[0][5], Wb[0][5], Wp[0][11],
        zb, h0f, h0b);

    // Layer 1 (x = h0, Kx = H)
    gate_rz_kernel<H_><<<32, 128, 0, stream>>>(h0b, h1b, h1f,
        Wb[1][0], Wp[1][1], Wb[1][1], Wp[1][3],
        Wb[1][3], Wp[1][7], Wb[1][4], Wp[1][9],
        rhb, zb);
    htilde_kernel<H_><<<32, 128, 0, stream>>>(h0b, rhb,
        Wb[1][2], Wp[1][5], Wb[1][5], Wp[1][11],
        zb, h1f, h1b);

    // Decoder
    decode_kernel<<<250, 128, 0, stream>>>(h1b, decWb, dec_b,
                                           out + (size_t)s * B_ * V_);
  }
}
